// LSTM_33225867002567
// MI455X (gfx1250) — compile-verified, matrix operand order checked by reading, not run on hardware
//
#include <hip/hip_runtime.h>
#include <cstdint>
#include <cstddef>

// ---------------------------------------------------------------------------
// LSTM encoder/decoder for MI455X (gfx1250), bf16 WMMA with fp32 accumulate.
//
// Shapes (fixed by setup_inputs): B=64, T=512, H=512, S=16, HOR=64.
// Total math ~0.6 TFLOP; weights ~6MB bf16 -> fully L2 resident (192MB L2).
// Problem is recurrence-bound: per-step GEMMs are (64|1024)x2048, K=512|1024.
// Strategy: v_wmma_f32_16x16x32_bf16 tiles; each wave computes the 4 gate
// tiles (i,f,g,o) of one 16x16 (row,hcol) block so the cell update is fully
// register-resident. Encoder: persistent 1-WG kernel, h double-buffered in
// LDS, c in VGPRs, one barrier per timestep; layer-1 stages its input slab
// into LDS with global_load_async_to_lds_b128 (+ s_wait_asynccnt).
// Decoder: one launch per dependency edge (launch = grid-wide sync).
// ---------------------------------------------------------------------------

typedef __attribute__((ext_vector_type(16))) __bf16 v16bf;
typedef __attribute__((ext_vector_type(8)))  float  v8f;

#define DEVINL __device__ __forceinline__

static constexpr int Hdim = 512;
static constexpr int FH   = 2048;   // 4*H
static constexpr int Tlen = 512;
static constexpr int Bb   = 64;
static constexpr int BS   = 1024;   // B*S
static constexpr int HOR  = 64;
static constexpr int KT   = Hdim / 32;  // 16 k-tiles of 32 per H
static constexpr int NTH  = Hdim / 16;  // 32 col-tiles of 16 per gate
static constexpr int HBYTES = Bb * Hdim * 2;  // one h slab in LDS (64KB)

DEVINL unsigned short f2b(float f) {            // f32 -> bf16 (RNE)
  union { float f; unsigned u; } v; v.f = f;
  unsigned r = (v.u + 0x7FFFu + ((v.u >> 16) & 1u)) >> 16;
  return (unsigned short)r;
}
DEVINL float b2f(unsigned short s) {
  union { unsigned u; float f; } v; v.u = ((unsigned)s) << 16; return v.f;
}
DEVINL float sigmoidf_(float x) { return 1.f / (1.f + __expf(-x)); }

DEVINL v8f wmma_bf16(v16bf a, v16bf b, v8f c) {
  // (neg_a, A, neg_b, B, c_mod, C, reuse_a, reuse_b)
  return __builtin_amdgcn_wmma_f32_16x16x32_bf16(false, a, false, b,
                                                 (short)0, c, false, false);
}

// A fragment: 16x32 bf16 tile from a row-major bf16 matrix (ldRow elements).
// Lane layout (ISA 7.12.2, 16-bit A): lane m=l&15 holds K offsets
//   lo-half (l<16): k = i (+8 for i>=8); hi-half: +8.
DEVINL v16bf load_a_frag(const unsigned short* __restrict__ base, int ldRow,
                         int rowBase, int kBase, int lane) {
  const int m = lane & 15, hi = lane >> 4;
  const unsigned short* p =
      base + (size_t)(rowBase + m) * ldRow + kBase + (hi ? 8 : 0);
  union { v16bf v; uint4 q[2]; } u;
  u.q[0] = *(const uint4*)(p);       // k .. k+7
  u.q[1] = *(const uint4*)(p + 16);  // k+16 .. k+23
  return u.v;
}

// B fragment from pre-packed weights: one contiguous 32B load per lane.
DEVINL v16bf load_b_frag(const unsigned short* __restrict__ P,
                         int nTile, int kTile, int lane) {
  const unsigned short* p =
      P + ((((size_t)nTile * KT + kTile) * 32 + lane) << 4);
  union { v16bf v; uint4 q[2]; } u;
  u.q[0] = *(const uint4*)(p);
  u.q[1] = *(const uint4*)(p + 8);
  return u.v;
}

// Async copy of one 16-byte chunk global -> LDS (CDNA5 ASYNCcnt path).
DEVINL void async_g2l_b128(unsigned lds_byte_off, const void* gptr) {
  unsigned long long ga = (unsigned long long)gptr;
  asm volatile("global_load_async_to_lds_b128 %0, %1, off"
               :: "v"(lds_byte_off), "v"(ga) : "memory");
}
DEVINL void async_wait0() {
  asm volatile("s_wait_asynccnt 0" ::: "memory");
}

// ---------------------------------------------------------------------------
// Pack W (FH x H, row-major f32) into B-fragment lane order, bf16.
// ---------------------------------------------------------------------------
__global__ void pack_w_kernel(const float* __restrict__ W,
                              unsigned short* __restrict__ P) {
  const int total = FH * Hdim;  // 2^20
  for (int idx = blockIdx.x * blockDim.x + threadIdx.x; idx < total;
       idx += gridDim.x * blockDim.x) {
    const int i     = idx & 15;
    const int lane  = (idx >> 4) & 31;
    const int kTile = (idx >> 9) & (KT - 1);
    const int nTile = idx >> 13;
    const int n  = nTile * 16 + (lane & 15);
    const int k  = kTile * 32 + i + ((i >= 8) ? 8 : 0) + ((lane >> 4) ? 8 : 0);
    P[idx] = f2b(W[(size_t)n * Hdim + k]);
  }
}

// ---------------------------------------------------------------------------
// Encoder layer 0: persistent single workgroup (32 waves), 512 timesteps.
// h (bf16) double-buffered in LDS; c lives in VGPRs across all steps.
// ---------------------------------------------------------------------------
__global__ __launch_bounds__(1024, 1) void enc_l0_kernel(
    const float* __restrict__ x, const float* __restrict__ wih0,
    const float* __restrict__ b0, const unsigned short* __restrict__ Whh,
    unsigned short* __restrict__ ys0, float* __restrict__ hF,
    float* __restrict__ cF) {
  extern __shared__ char smem[];           // 2 * 64KB h buffers
  unsigned short* hA = (unsigned short*)smem;
  unsigned short* hB = (unsigned short*)(smem + HBYTES);
  const int tid = threadIdx.x, wave = tid >> 5, lane = tid & 31;
  const int m = lane & 15, hi = lane >> 4;

  for (int i = tid; i < Bb * Hdim; i += 1024) hA[i] = 0;
  float creg[4][8];
  // Hoist per-task bias / input-weight rows out of the time loop.
  float bIs[4], bFs[4], bGs[4], bOs[4], wIs[4], wFs[4], wGs[4], wOs[4];
#pragma unroll
  for (int q = 0; q < 4; ++q) {
    const int task = wave * 4 + q;
    const int nH = (task & 31) * 16 + m;
    bIs[q] = b0[nH];            bFs[q] = b0[nH + Hdim];
    bGs[q] = b0[nH + 2 * Hdim]; bOs[q] = b0[nH + 3 * Hdim];
    wIs[q] = wih0[nH];            wFs[q] = wih0[nH + Hdim];
    wGs[q] = wih0[nH + 2 * Hdim]; wOs[q] = wih0[nH + 3 * Hdim];
#pragma unroll
    for (int r = 0; r < 8; ++r) creg[q][r] = 0.f;
  }
  __syncthreads();

  int cur = 0;
  for (int t = 0; t < Tlen; ++t) {
    const unsigned short* hc = cur ? hB : hA;
    unsigned short*       hn = cur ? hA : hB;
#pragma unroll
    for (int q = 0; q < 4; ++q) {
      const int task = wave * 4 + q;            // 128 tasks: 4 rt x 32 hc
      const int rt = task >> 5, hcid = task & 31;
      const int rowBase = rt * 16, nH = hcid * 16 + m;
      v8f aI{}, aF{}, aG{}, aO{};
#pragma unroll
      for (int r = 0; r < 8; ++r) {
        const int row = rowBase + hi * 8 + r;
        const float u = x[(size_t)row * Tlen + t];
        aI[r] = bIs[q] + u * wIs[q]; aF[r] = bFs[q] + u * wFs[q];
        aG[r] = bGs[q] + u * wGs[q]; aO[r] = bOs[q] + u * wOs[q];
      }
#pragma unroll 4
      for (int kt = 0; kt < KT; ++kt) {
        if (kt + 1 < KT)  // keep next weight k-tile near the WGP
          __builtin_prefetch(
              Whh + ((((size_t)hcid * KT + (kt + 1)) * 32 + lane) << 4), 0, 3);
        const v16bf a = load_a_frag(hc, Hdim, rowBase, kt * 32, lane);
        aI = wmma_bf16(a, load_b_frag(Whh, 0 * NTH + hcid, kt, lane), aI);
        aF = wmma_bf16(a, load_b_frag(Whh, 1 * NTH + hcid, kt, lane), aF);
        aG = wmma_bf16(a, load_b_frag(Whh, 2 * NTH + hcid, kt, lane), aG);
        aO = wmma_bf16(a, load_b_frag(Whh, 3 * NTH + hcid, kt, lane), aO);
      }
#pragma unroll
      for (int r = 0; r < 8; ++r) {
        const float ig = sigmoidf_(aI[r]), fg = sigmoidf_(aF[r]);
        const float gg = tanhf(aG[r]),     og = sigmoidf_(aO[r]);
        const float c = fg * creg[q][r] + ig * gg;
        creg[q][r] = c;
        const float h = og * tanhf(c);
        const int row = rowBase + hi * 8 + r;
        const unsigned short hb16 = f2b(h);
        hn[row * Hdim + nH] = hb16;
        ys0[(size_t)t * Bb * Hdim + (size_t)row * Hdim + nH] = hb16;
        if (t == Tlen - 1) {
          hF[(size_t)row * Hdim + nH] = h;
          cF[(size_t)row * Hdim + nH] = c;
        }
      }
    }
    __builtin_amdgcn_s_cluster_barrier();  // NOP when not in a cluster
    __syncthreads();                       // h(t+1) visible, reads of h(t) done
    cur ^= 1;
  }
}

// ---------------------------------------------------------------------------
// Encoder layer 1: gates = ys0_t@wih1^T + h@whh1^T + b1.
// ys0_t slab (64KB) staged into LDS via async global->LDS copies.
// ---------------------------------------------------------------------------
__global__ __launch_bounds__(1024, 1) void enc_l1_kernel(
    const unsigned short* __restrict__ ys0, const float* __restrict__ b1,
    const unsigned short* __restrict__ Wih, const unsigned short* __restrict__ Whh,
    float* __restrict__ hF, float* __restrict__ cF) {
  extern __shared__ char smem[];  // [0,128KB): h double buffer, [128KB,192KB): x
  unsigned short* hA = (unsigned short*)smem;
  unsigned short* hB = (unsigned short*)(smem + HBYTES);
  unsigned short* xbuf = (unsigned short*)(smem + 2 * HBYTES);
  const unsigned xbase = 2u * (unsigned)HBYTES;  // LDS byte offset of xbuf
  const int tid = threadIdx.x, wave = tid >> 5, lane = tid & 31;
  const int m = lane & 15, hi = lane >> 4;

  for (int i = tid; i < Bb * Hdim; i += 1024) hA[i] = 0;
  float creg[4][8];
  float bIs[4], bFs[4], bGs[4], bOs[4];
#pragma unroll
  for (int q = 0; q < 4; ++q) {
    const int nH = ((wave * 4 + q) & 31) * 16 + m;
    bIs[q] = b1[nH];            bFs[q] = b1[nH + Hdim];
    bGs[q] = b1[nH + 2 * Hdim]; bOs[q] = b1[nH + 3 * Hdim];
#pragma unroll
    for (int r = 0; r < 8; ++r) creg[q][r] = 0.f;
  }
  __syncthreads();

  int cur = 0;
  for (int t = 0; t < Tlen; ++t) {
    // ---- async-stage x_t = ys0[t] (64KB) into LDS (ASYNCcnt path) ----
    {
      const unsigned short* src = ys0 + (size_t)t * Bb * Hdim;
#pragma unroll
      for (int c = 0; c < 4; ++c) {                 // 4096 chunks / 1024 thr
        const int chunk = tid + c * 1024;           // 16B per chunk
        async_g2l_b128(xbase + (unsigned)chunk * 16u, src + chunk * 8);
      }
      async_wait0();
      __syncthreads();  // xbuf ready for every wave
    }
    const unsigned short* hc = cur ? hB : hA;
    unsigned short*       hn = cur ? hA : hB;
#pragma unroll
    for (int q = 0; q < 4; ++q) {
      const int task = wave * 4 + q;
      const int rt = task >> 5, hcid = task & 31;
      const int rowBase = rt * 16, nH = hcid * 16 + m;
      v8f aI{}, aF{}, aG{}, aO{};
#pragma unroll
      for (int r = 0; r < 8; ++r) {
        aI[r] = bIs[q]; aF[r] = bFs[q]; aG[r] = bGs[q]; aO[r] = bOs[q];
      }
#pragma unroll 4
      for (int kt = 0; kt < KT; ++kt) {   // input contribution (LDS x)
        const v16bf a = load_a_frag(xbuf, Hdim, rowBase, kt * 32, lane);
        aI = wmma_bf16(a, load_b_frag(Wih, 0 * NTH + hcid, kt, lane), aI);
        aF = wmma_bf16(a, load_b_frag(Wih, 1 * NTH + hcid, kt, lane), aF);
        aG = wmma_bf16(a, load_b_frag(Wih, 2 * NTH + hcid, kt, lane), aG);
        aO = wmma_bf16(a, load_b_frag(Wih, 3 * NTH + hcid, kt, lane), aO);
      }
#pragma unroll 4
      for (int kt = 0; kt < KT; ++kt) {   // recurrent contribution (LDS h)
        if (kt + 1 < KT)
          __builtin_prefetch(
              Whh + ((((size_t)hcid * KT + (kt + 1)) * 32 + lane) << 4), 0, 3);
        const v16bf a = load_a_frag(hc, Hdim, rowBase, kt * 32, lane);
        aI = wmma_bf16(a, load_b_frag(Whh, 0 * NTH + hcid, kt, lane), aI);
        aF = wmma_bf16(a, load_b_frag(Whh, 1 * NTH + hcid, kt, lane), aF);
        aG = wmma_bf16(a, load_b_frag(Whh, 2 * NTH + hcid, kt, lane), aG);
        aO = wmma_bf16(a, load_b_frag(Whh, 3 * NTH + hcid, kt, lane), aO);
      }
#pragma unroll
      for (int r = 0; r < 8; ++r) {
        const float ig = sigmoidf_(aI[r]), fg = sigmoidf_(aF[r]);
        const float gg = tanhf(aG[r]),     og = sigmoidf_(aO[r]);
        const float c = fg * creg[q][r] + ig * gg;
        creg[q][r] = c;
        const float h = og * tanhf(c);
        const int row = rowBase + hi * 8 + r;
        hn[row * Hdim + nH] = f2b(h);
        if (t == Tlen - 1) {
          hF[(size_t)row * Hdim + nH] = h;
          cF[(size_t)row * Hdim + nH] = c;
        }
      }
    }
    __builtin_amdgcn_s_cluster_barrier();
    __syncthreads();
    cur ^= 1;
  }
}

// ---------------------------------------------------------------------------
// Replicate encoder final states over S samples; seed u with init_input.
// ---------------------------------------------------------------------------
__global__ void rep_init_kernel(
    const float* __restrict__ h0f, const float* __restrict__ c0f,
    const float* __restrict__ h1f, const float* __restrict__ c1f,
    const float* __restrict__ init_input, unsigned short* __restrict__ h0b,
    float* __restrict__ c0, unsigned short* __restrict__ h1b,
    float* __restrict__ c1, float* __restrict__ ubuf) {
  const int total = BS * Hdim;
  for (int idx = blockIdx.x * blockDim.x + threadIdx.x; idx < total;
       idx += gridDim.x * blockDim.x) {
    const int r = idx >> 9, k = idx & 511;
    const size_t s = (size_t)(r >> 4) * Hdim + k;   // source batch row
    h0b[idx] = f2b(h0f[s]);
    c0[idx]  = c0f[s];
    h1b[idx] = f2b(h1f[s]);
    c1[idx]  = c1f[s];
    if (k == 0) ubuf[r] = init_input[r];
  }
}

// ---------------------------------------------------------------------------
// Decoder layer 0: 1024x2048 gates, K=512. 64 WGs x 8 waves, 4 tasks/wave.
// ---------------------------------------------------------------------------
__global__ __launch_bounds__(256) void dec_l0_kernel(
    const float* __restrict__ ubuf, const float* __restrict__ wih0,
    const float* __restrict__ b0, const unsigned short* __restrict__ Whh,
    const unsigned short* __restrict__ hIn, unsigned short* __restrict__ hOut,
    float* __restrict__ cSt) {
  const int tid = threadIdx.x, wave = tid >> 5, lane = tid & 31;
  const int m = lane & 15, hi = lane >> 4;
  const int gw = blockIdx.x * 8 + wave;  // 0..511
#pragma unroll
  for (int q = 0; q < 4; ++q) {
    const int task = gw * 4 + q;         // 2048 tasks: 64 rt x 32 hc
    const int rt = task >> 5, hc = task & 31;
    const int rowBase = rt * 16, nH = hc * 16 + m;
    v8f aI{}, aF{}, aG{}, aO{};
    const float bI = b0[nH], bFv = b0[nH + Hdim];
    const float bG = b0[nH + 2 * Hdim], bO = b0[nH + 3 * Hdim];
    const float wI = wih0[nH], wF = wih0[nH + Hdim];
    const float wG = wih0[nH + 2 * Hdim], wO = wih0[nH + 3 * Hdim];
#pragma unroll
    for (int r = 0; r < 8; ++r) {
      const int row = rowBase + hi * 8 + r;
      const float u = ubuf[row];
      aI[r] = bI + u * wI; aF[r] = bFv + u * wF;
      aG[r] = bG + u * wG; aO[r] = bO + u * wO;
    }
#pragma unroll 4
    for (int kt = 0; kt < KT; ++kt) {
      if (kt + 1 < KT)
        __builtin_prefetch(
            Whh + ((((size_t)hc * KT + (kt + 1)) * 32 + lane) << 4), 0, 3);
      const v16bf a = load_a_frag(hIn, Hdim, rowBase, kt * 32, lane);
      aI = wmma_bf16(a, load_b_frag(Whh, 0 * NTH + hc, kt, lane), aI);
      aF = wmma_bf16(a, load_b_frag(Whh, 1 * NTH + hc, kt, lane), aF);
      aG = wmma_bf16(a, load_b_frag(Whh, 2 * NTH + hc, kt, lane), aG);
      aO = wmma_bf16(a, load_b_frag(Whh, 3 * NTH + hc, kt, lane), aO);
    }
#pragma unroll
    for (int r = 0; r < 8; ++r) {
      const int row = rowBase + hi * 8 + r;
      const size_t p = (size_t)row * Hdim + nH;
      const float ig = sigmoidf_(aI[r]), fg = sigmoidf_(aF[r]);
      const float gg = tanhf(aG[r]),     og = sigmoidf_(aO[r]);
      const float c = fg * cSt[p] + ig * gg;
      cSt[p] = c;
      hOut[p] = f2b(og * tanhf(c));
    }
  }
}

// ---------------------------------------------------------------------------
// Decoder layer 1: gates = h0_new@wih1^T + h1@whh1^T + b1 (K = 512 + 512).
// ---------------------------------------------------------------------------
__global__ __launch_bounds__(256) void dec_l1_kernel(
    const float* __restrict__ b1, const unsigned short* __restrict__ Wih,
    const unsigned short* __restrict__ Whh,
    const unsigned short* __restrict__ h0New,
    const unsigned short* __restrict__ h1In,
    unsigned short* __restrict__ h1Out, float* __restrict__ cSt) {
  const int tid = threadIdx.x, wave = tid >> 5, lane = tid & 31;
  const int m = lane & 15, hi = lane >> 4;
  const int gw = blockIdx.x * 8 + wave;
#pragma unroll
  for (int q = 0; q < 4; ++q) {
    const int task = gw * 4 + q;
    const int rt = task >> 5, hc = task & 31;
    const int rowBase = rt * 16, nH = hc * 16 + m;
    v8f aI{}, aF{}, aG{}, aO{};
    const float bI = b1[nH], bFv = b1[nH + Hdim];
    const float bG = b1[nH + 2 * Hdim], bO = b1[nH + 3 * Hdim];
#pragma unroll
    for (int r = 0; r < 8; ++r) { aI[r] = bI; aF[r] = bFv; aG[r] = bG; aO[r] = bO; }
#pragma unroll 4
    for (int kt = 0; kt < KT; ++kt) {
      const v16bf a = load_a_frag(h0New, Hdim, rowBase, kt * 32, lane);
      aI = wmma_bf16(a, load_b_frag(Wih, 0 * NTH + hc, kt, lane), aI);
      aF = wmma_bf16(a, load_b_frag(Wih, 1 * NTH + hc, kt, lane), aF);
      aG = wmma_bf16(a, load_b_frag(Wih, 2 * NTH + hc, kt, lane), aG);
      aO = wmma_bf16(a, load_b_frag(Wih, 3 * NTH + hc, kt, lane), aO);
    }
#pragma unroll 4
    for (int kt = 0; kt < KT; ++kt) {
      if (kt + 1 < KT)
        __builtin_prefetch(
            Whh + ((((size_t)hc * KT + (kt + 1)) * 32 + lane) << 4), 0, 3);
      const v16bf a = load_a_frag(h1In, Hdim, rowBase, kt * 32, lane);
      aI = wmma_bf16(a, load_b_frag(Whh, 0 * NTH + hc, kt, lane), aI);
      aF = wmma_bf16(a, load_b_frag(Whh, 1 * NTH + hc, kt, lane), aF);
      aG = wmma_bf16(a, load_b_frag(Whh, 2 * NTH + hc, kt, lane), aG);
      aO = wmma_bf16(a, load_b_frag(Whh, 3 * NTH + hc, kt, lane), aO);
    }
#pragma unroll
    for (int r = 0; r < 8; ++r) {
      const int row = rowBase + hi * 8 + r;
      const size_t p = (size_t)row * Hdim + nH;
      const float ig = sigmoidf_(aI[r]), fg = sigmoidf_(aF[r]);
      const float gg = tanhf(aG[r]),     og = sigmoidf_(aO[r]);
      const float c = fg * cSt[p] + ig * gg;
      cSt[p] = c;
      h1Out[p] = f2b(og * tanhf(c));
    }
  }
}

// ---------------------------------------------------------------------------
// Readout: pred = h1 @ w_out^T + b_out ; feeds next step's u and the output.
// ---------------------------------------------------------------------------
__global__ __launch_bounds__(256) void dec_pred_kernel(
    const unsigned short* __restrict__ h1, const float* __restrict__ wout,
    const float* __restrict__ bout, float* __restrict__ ubuf,
    float* __restrict__ out, int t) {
  const int r = blockIdx.x * blockDim.x + threadIdx.x;
  if (r >= BS) return;
  float acc = bout[0];
  const unsigned short* hp = h1 + (size_t)r * Hdim;
  for (int k = 0; k < Hdim; k += 8) {
    union { uint4 q; unsigned short u[8]; } hv;
    hv.q = *(const uint4*)(hp + k);
    const float4 w0 = *(const float4*)(wout + k);
    const float4 w1 = *(const float4*)(wout + k + 4);
    acc += b2f(hv.u[0]) * w0.x + b2f(hv.u[1]) * w0.y +
           b2f(hv.u[2]) * w0.z + b2f(hv.u[3]) * w0.w +
           b2f(hv.u[4]) * w1.x + b2f(hv.u[5]) * w1.y +
           b2f(hv.u[6]) * w1.z + b2f(hv.u[7]) * w1.w;
  }
  ubuf[r] = acc;
  out[(size_t)r * HOR + t] = acc;  // (B,S,HOR) row-major == r*HOR + t
}

// ---------------------------------------------------------------------------
extern "C" void kernel_launch(void* const* d_in, const int* in_sizes, int n_in,
                              void* d_out, int out_size, void* d_ws,
                              size_t ws_size, hipStream_t stream) {
  (void)in_sizes; (void)n_in; (void)out_size; (void)ws_size;
  const float* x    = (const float*)d_in[0];
  const float* init = (const float*)d_in[1];
  const float* wih0 = (const float*)d_in[2];
  const float* whh0 = (const float*)d_in[3];
  const float* b0   = (const float*)d_in[4];
  const float* wih1 = (const float*)d_in[5];
  const float* whh1 = (const float*)d_in[6];
  const float* b1   = (const float*)d_in[7];
  const float* wout = (const float*)d_in[8];
  const float* bout = (const float*)d_in[9];
  float* out = (float*)d_out;

  char* ws = (char*)d_ws;
  size_t off = 0;
  auto carve = [&](size_t bytes) -> char* {
    char* p = ws + off;
    off += (bytes + 255) & ~(size_t)255;
    return p;
  };
  unsigned short* whh0p = (unsigned short*)carve((size_t)FH * Hdim * 2);
  unsigned short* wih1p = (unsigned short*)carve((size_t)FH * Hdim * 2);
  unsigned short* whh1p = (unsigned short*)carve((size_t)FH * Hdim * 2);
  unsigned short* ys0   = (unsigned short*)carve((size_t)Tlen * Bb * Hdim * 2);
  float* h0f = (float*)carve((size_t)Bb * Hdim * 4);
  float* c0f = (float*)carve((size_t)Bb * Hdim * 4);
  float* h1f = (float*)carve((size_t)Bb * Hdim * 4);
  float* c1f = (float*)carve((size_t)Bb * Hdim * 4);
  unsigned short* h0a = (unsigned short*)carve((size_t)BS * Hdim * 2);
  unsigned short* h0b = (unsigned short*)carve((size_t)BS * Hdim * 2);
  unsigned short* h1a = (unsigned short*)carve((size_t)BS * Hdim * 2);
  unsigned short* h1b = (unsigned short*)carve((size_t)BS * Hdim * 2);
  float* c0   = (float*)carve((size_t)BS * Hdim * 4);
  float* c1   = (float*)carve((size_t)BS * Hdim * 4);
  float* ubuf = (float*)carve((size_t)BS * 4);

  // Pack recurrent/input weights into WMMA B-fragment layout (bf16).
  pack_w_kernel<<<256, 256, 0, stream>>>(whh0, whh0p);
  pack_w_kernel<<<256, 256, 0, stream>>>(wih1, wih1p);
  pack_w_kernel<<<256, 256, 0, stream>>>(whh1, whh1p);

  // Encoder: two persistent single-WG scans (barrier per timestep).
  enc_l0_kernel<<<1, 1024, 2 * HBYTES, stream>>>(x, wih0, b0, whh0p, ys0,
                                                 h0f, c0f);
  enc_l1_kernel<<<1, 1024, 3 * HBYTES, stream>>>(ys0, b1, wih1p, whh1p,
                                                 h1f, c1f);

  // Expand states over samples; seed u.
  rep_init_kernel<<<256, 256, 0, stream>>>(h0f, c0f, h1f, c1f, init, h0a, c0,
                                           h1a, c1, ubuf);

  // Autoregressive decode: launches provide the grid-wide sync.
  unsigned short *h0r = h0a, *h0w = h0b, *h1r = h1a, *h1w = h1b;
  for (int t = 0; t < HOR; ++t) {
    dec_l0_kernel<<<64, 256, 0, stream>>>(ubuf, wih0, b0, whh0p, h0r, h0w, c0);
    dec_l1_kernel<<<64, 256, 0, stream>>>(b1, wih1p, whh1p, h0w, h1r, h1w, c1);
    dec_pred_kernel<<<4, 256, 0, stream>>>(h1w, wout, bout, ubuf, out, t);
    unsigned short* tmp;
    tmp = h0r; h0r = h0w; h0w = tmp;
    tmp = h1r; h1r = h1w; h1w = tmp;
  }
}
